// PredictiveCodingRNN_81913616269693
// MI455X (gfx1250) — matrix operand order repaired and checked by reading, not compiled, not run on hardware
//
#include <hip/hip_runtime.h>
#include <hip/hip_bf16.h>
#include <stdint.h>

// ---------------------------------------------------------------------------
// PredictiveCodingRNN on MI455X (gfx1250, wave32, WMMA)
// B=32, S=256, H=V=K=2048.  bf16 WMMA 16x16x32, f32 accumulate.
//  - Weights packed once per call into WMMA-fragment-native bf16 layout;
//    58.7MB resident in the 192MB L2 across all 256 steps.
//  - Each wave owns M=32 (both 16-row tiles): B fragments loaded ONCE.
//  - A operand (packed e_t / h) staged into LDS with double-buffered
//    global_load_async_to_lds_b128 + s_wait_asynccnt (ASYNCcnt pipeline).
// ---------------------------------------------------------------------------

#define B_   32
#define S_   256
#define H_   2048
#define V_   2048
#define K_   2048      // K dim of every GEMM in this model
#define NT   64        // output-column tile per workgroup (4 waves x 16)
#define KB_  64        // K-slices of 32 (K_/32)
#define KCH  4         // K chunks for LDS double buffering
#define KBC  16        // K-slices per chunk (KB_/KCH)

typedef __attribute__((ext_vector_type(16))) __bf16 v16bf;
typedef __attribute__((ext_vector_type(8)))  float  v8f;

// round-to-nearest-even f32 -> bf16 (bit pattern as u16)
__device__ __forceinline__ uint16_t f2bf(float f) {
    union { float f; uint32_t u; } v; v.f = f;
    uint32_t u = v.u + 0x7fffu + ((v.u >> 16) & 1u);
    return (uint16_t)(u >> 16);
}

// ---- CDNA5 async copy: global -> LDS, tracked by ASYNCcnt -----------------
__device__ __forceinline__ void async_ld16(uint32_t lds_byte, const void* g) {
    asm volatile("global_load_async_to_lds_b128 %0, %1, off"
                 :: "v"(lds_byte), "v"(g) : "memory");
}
template <int N>
__device__ __forceinline__ void wait_async() {
    asm volatile("s_wait_asynccnt %0" :: "n"(N) : "memory");
}

// ---- packed-fragment index math ------------------------------------------
// A-matrix (MxK, 16-bit) 16x32 tile layout (ISA 7.12.2):
//   lanes 0-15  : VGPR0..3 = K 0..7 (pairs), VGPR4..7 = K 16..23
//   lanes 16-31 : VGPR0..3 = K 8..15,        VGPR4..7 = K 24..31
// Each tile is a 512-bf16 chunk; lane L reads 16 contiguous bf16 (32B).
__device__ __forceinline__ long packA_idx(int m, int k, int K) {
    long chunk = (long)(m >> 4) * (K >> 5) + (k >> 5);
    int ko   = k & 31;
    int half = (ko >> 3) & 1;                  // 0,1,0,1 per 8-group
    int j    = (ko & 7) + ((ko >> 4) << 3);    // +8 for K%32 >= 16
    int lane = (half << 4) + (m & 15);
    return chunk * 512 + lane * 16 + j;
}

// B-matrix (KxN = 32x16, 16-bit): lane L holds column n=L%16,
// K = (L/16)*8 + 0..7 then 16 + (L/16)*8 + 0..7 (same order as A).
__global__ void pack_w(const float* __restrict__ W, uint16_t* __restrict__ out,
                       int Ntot, int K) {
    long e = (long)blockIdx.x * blockDim.x + threadIdx.x;
    long total = (long)Ntot * K;
    if (e >= total) return;
    long chunk = e >> 9;                 // /512
    int  lane  = (int)((e >> 4) & 31);
    int  j     = (int)(e & 15);
    int  KB    = K >> 5;
    int  kb    = (int)(chunk % KB);
    int  nt    = (int)(chunk / KB);
    int  n     = (nt << 4) + (lane & 15);
    int  k     = (kb << 5) + ((lane >> 4) << 3) + (j & 7) + ((j >> 3) << 4);
    out[e] = f2bf(W[(long)n * K + k]);
}

// ---- h0 -> (f32 state, packed bf16 A-fragments) ---------------------------
__global__ void init_h(const float* __restrict__ h0,
                       float* __restrict__ hf, uint16_t* __restrict__ hp) {
    int e = blockIdx.x * blockDim.x + threadIdx.x;   // 0 .. B*H-1
    int m = e >> 11;
    int k = e & (H_ - 1);
    float v = h0[e];
    hf[e] = v;
    hp[packA_idx(m, k, K_)] = f2bf(v);
}

// ---- softmax + error: e_t = onehot(x_t) - softmax(x_hat) ------------------
__global__ void __launch_bounds__(256)
softmax_err(const float* __restrict__ xhat, const long long* __restrict__ x,
            int t, float* __restrict__ errors, uint16_t* __restrict__ e_pack) {
    __shared__ float red[256];
    int b = blockIdx.x, tid = threadIdx.x;
    const float* row = xhat + (long)b * V_;
    float vals[8];
    float mx = -3.4e38f;
#pragma unroll
    for (int i = 0; i < 8; ++i) { vals[i] = row[tid + i * 256]; mx = fmaxf(mx, vals[i]); }
    red[tid] = mx; __syncthreads();
    for (int s = 128; s > 0; s >>= 1) { if (tid < s) red[tid] = fmaxf(red[tid], red[tid + s]); __syncthreads(); }
    mx = red[0]; __syncthreads();
    float sum = 0.f;
#pragma unroll
    for (int i = 0; i < 8; ++i) { vals[i] = __expf(vals[i] - mx); sum += vals[i]; }
    red[tid] = sum; __syncthreads();
    for (int s = 128; s > 0; s >>= 1) { if (tid < s) red[tid] += red[tid + s]; __syncthreads(); }
    float inv = 1.0f / red[0];
    int tgt = (int)x[(long)b * S_ + t];
    float* err_row = errors + ((long)b * S_ + t) * V_;
#pragma unroll
    for (int i = 0; i < 8; ++i) {
        int v = tid + i * 256;
        float e = ((v == tgt) ? 1.0f : 0.0f) - vals[i] * inv;
        err_row[v] = e;
        e_pack[packA_idx(b, v, K_)] = f2bf(e);
    }
}

// ---- stage one K-chunk of A panels into LDS (async DMA) -------------------
// NREG regions of KBC*1024 bytes each; region r source = srcs[r], dest =
// lds_base + r*16384.  128 threads, 16B units, 8 asyncs/thread/region.
template <int NREG>
__device__ __forceinline__ void stage_A(const char* const (&srcs)[NREG],
                                        uint32_t lds_base, int tid) {
#pragma unroll
    for (int r = 0; r < NREG; ++r) {
        uint32_t dst = lds_base + r * (KBC * 1024);
#pragma unroll
        for (int j = 0; j < 8; ++j) {
            int off = (tid + j * 128) * 16;
            async_ld16(dst + off, srcs[r] + off);
        }
    }
}

// ---- fused GRU step -------------------------------------------------------
// grid = H/NT (32) blocks of 128 threads (4 waves); wave = one 16-col tile,
// BOTH 16-row tiles (M=32) -> each B fragment loaded once, 12 WMMAs/K-slice.
// A panels (e_pack, h_pack, both row tiles) double-buffered in LDS via
// async DMA: 2 x 64KB buffers, 4 K-chunks of 512.
__global__ void __launch_bounds__(128)
gru_step(const uint16_t* __restrict__ e_pack,
         const uint16_t* __restrict__ h_pack_in,
         const float*    __restrict__ h_in,
         const uint16_t* __restrict__ Wih_p,
         const uint16_t* __restrict__ Whh_p,
         const float*    __restrict__ b_ih,
         const float*    __restrict__ b_hh,
         float*          __restrict__ h_out,
         uint16_t*       __restrict__ h_pack_out) {
    extern __shared__ char smem[];                    // 2 * 64KB
    int tid  = threadIdx.x;
    int wave = tid >> 5;
    int lane = tid & 31;
    int c0   = blockIdx.x * NT + wave * 16;           // column within H

    const v16bf* BiP = (const v16bf*)Wih_p;
    const v16bf* BhP = (const v16bf*)Whh_p;
    long ntr = (long)((0 * H_ + c0) >> 4) * KB_;      // B chunk bases per gate
    long ntz = (long)((1 * H_ + c0) >> 4) * KB_;
    long ntn = (long)((2 * H_ + c0) >> 4) * KB_;

    uint32_t lds0 = (uint32_t)(uintptr_t)&smem[0];

    v8f ar0 = {}, ar1 = {}, az0 = {}, az1 = {};
    v8f ai0 = {}, ai1 = {}, ah0a = {}, ah1a = {};

    // chunk c sources: e rt0, e rt1, h rt0, h rt1 (16KB each, contiguous)
    {
        const char* s0[4] = {
            (const char*)e_pack + (0 * KB_ + 0 * KBC) * 1024,
            (const char*)e_pack + (1 * KB_ + 0 * KBC) * 1024,
            (const char*)h_pack_in + (0 * KB_ + 0 * KBC) * 1024,
            (const char*)h_pack_in + (1 * KB_ + 0 * KBC) * 1024 };
        stage_A<4>(s0, lds0, tid);
    }
    for (int c = 0; c < KCH; ++c) {
        if (c + 1 < KCH) {
            const char* sn[4] = {
                (const char*)e_pack + (0 * KB_ + (c + 1) * KBC) * 1024,
                (const char*)e_pack + (1 * KB_ + (c + 1) * KBC) * 1024,
                (const char*)h_pack_in + (0 * KB_ + (c + 1) * KBC) * 1024,
                (const char*)h_pack_in + (1 * KB_ + (c + 1) * KBC) * 1024 };
            stage_A<4>(sn, lds0 + ((c + 1) & 1) * 65536, tid);
            wait_async<32>();          // current chunk landed (in-order)
        } else {
            wait_async<0>();
        }
        __syncthreads();
        uint32_t bb = lds0 + (c & 1) * 65536;
        for (int kb = 0; kb < KBC; ++kb) {
            uint32_t ao = bb + kb * 1024 + lane * 32;
            v16bf ae0 = *(const v16bf*)(smem + (ao - lds0));            // e rt0
            v16bf ae1 = *(const v16bf*)(smem + (ao - lds0) + 16384);    // e rt1
            v16bf ah0 = *(const v16bf*)(smem + (ao - lds0) + 32768);    // h rt0
            v16bf ah1 = *(const v16bf*)(smem + (ao - lds0) + 49152);    // h rt1
            long kk = c * KBC + kb;
            v16bf bir = BiP[(ntr + kk) * 32 + lane];
            v16bf bhr = BhP[(ntr + kk) * 32 + lane];
            v16bf biz = BiP[(ntz + kk) * 32 + lane];
            v16bf bhz = BhP[(ntz + kk) * 32 + lane];
            v16bf bin = BiP[(ntn + kk) * 32 + lane];
            v16bf bhn = BhP[(ntn + kk) * 32 + lane];
            ar0  = __builtin_amdgcn_wmma_f32_16x16x32_bf16(false, ae0, false, bir, (short)0, ar0,  false, false);
            ar1  = __builtin_amdgcn_wmma_f32_16x16x32_bf16(false, ae1, false, bir, (short)0, ar1,  false, false);
            ar0  = __builtin_amdgcn_wmma_f32_16x16x32_bf16(false, ah0, false, bhr, (short)0, ar0,  false, false);
            ar1  = __builtin_amdgcn_wmma_f32_16x16x32_bf16(false, ah1, false, bhr, (short)0, ar1,  false, false);
            az0  = __builtin_amdgcn_wmma_f32_16x16x32_bf16(false, ae0, false, biz, (short)0, az0,  false, false);
            az1  = __builtin_amdgcn_wmma_f32_16x16x32_bf16(false, ae1, false, biz, (short)0, az1,  false, false);
            az0  = __builtin_amdgcn_wmma_f32_16x16x32_bf16(false, ah0, false, bhz, (short)0, az0,  false, false);
            az1  = __builtin_amdgcn_wmma_f32_16x16x32_bf16(false, ah1, false, bhz, (short)0, az1,  false, false);
            ai0  = __builtin_amdgcn_wmma_f32_16x16x32_bf16(false, ae0, false, bin, (short)0, ai0,  false, false);
            ai1  = __builtin_amdgcn_wmma_f32_16x16x32_bf16(false, ae1, false, bin, (short)0, ai1,  false, false);
            ah0a = __builtin_amdgcn_wmma_f32_16x16x32_bf16(false, ah0, false, bhn, (short)0, ah0a, false, false);
            ah1a = __builtin_amdgcn_wmma_f32_16x16x32_bf16(false, ah1, false, bhn, (short)0, ah1a, false, false);
        }
        __syncthreads();               // protect buffer before next prefetch
    }

    int col = c0 + (lane & 15);
    float br   = b_ih[col]       + b_hh[col];
    float bz   = b_ih[H_ + col]  + b_hh[H_ + col];
    float bi_n = b_ih[2 * H_ + col];
    float bh_n = b_hh[2 * H_ + col];
#pragma unroll
    for (int i = 0; i < 8; ++i) {
        int mrow = i + ((lane >> 4) << 3);
        {   // row tile 0
            int m = mrow;
            float r = 1.0f / (1.0f + __expf(-(ar0[i] + br)));
            float z = 1.0f / (1.0f + __expf(-(az0[i] + bz)));
            float n = tanhf(ai0[i] + bi_n + r * (ah0a[i] + bh_n));
            float hp = h_in[(long)m * H_ + col];
            float hn = (1.0f - z) * n + z * hp;
            h_out[(long)m * H_ + col] = hn;
            h_pack_out[packA_idx(m, col, K_)] = f2bf(hn);
        }
        {   // row tile 1
            int m = 16 + mrow;
            float r = 1.0f / (1.0f + __expf(-(ar1[i] + br)));
            float z = 1.0f / (1.0f + __expf(-(az1[i] + bz)));
            float n = tanhf(ai1[i] + bi_n + r * (ah1a[i] + bh_n));
            float hp = h_in[(long)m * H_ + col];
            float hn = (1.0f - z) * n + z * hp;
            h_out[(long)m * H_ + col] = hn;
            h_pack_out[packA_idx(m, col, K_)] = f2bf(hn);
        }
    }
}

// ---- projection: x_hat = h @ W_p^T + b_p ----------------------------------
// Same structure: 4 waves x 16 cols, M=32 per wave, LDS-staged A (2x32KB).
__global__ void __launch_bounds__(128)
proj_step(const uint16_t* __restrict__ h_pack,
          const uint16_t* __restrict__ Wp_p,
          const float*    __restrict__ b_p,
          float*          __restrict__ xhat,
          float*          __restrict__ outputs,   // (B,S,V) base; null = skip
          int t) {
    extern __shared__ char smem[];                // 2 * 32KB
    int tid  = threadIdx.x;
    int wave = tid >> 5;
    int lane = tid & 31;
    int c0   = blockIdx.x * NT + wave * 16;
    const v16bf* Bp = (const v16bf*)Wp_p;
    long nb = (long)(c0 >> 4) * KB_;
    uint32_t lds0 = (uint32_t)(uintptr_t)&smem[0];

    v8f acc0 = {}, acc1 = {};
    {
        const char* s0[2] = {
            (const char*)h_pack + (0 * KB_ + 0 * KBC) * 1024,
            (const char*)h_pack + (1 * KB_ + 0 * KBC) * 1024 };
        stage_A<2>(s0, lds0, tid);
    }
    for (int c = 0; c < KCH; ++c) {
        if (c + 1 < KCH) {
            const char* sn[2] = {
                (const char*)h_pack + (0 * KB_ + (c + 1) * KBC) * 1024,
                (const char*)h_pack + (1 * KB_ + (c + 1) * KBC) * 1024 };
            stage_A<2>(sn, lds0 + ((c + 1) & 1) * 32768, tid);
            wait_async<16>();
        } else {
            wait_async<0>();
        }
        __syncthreads();
        uint32_t bb = (uint32_t)((c & 1) * 32768);
        for (int kb = 0; kb < KBC; ++kb) {
            uint32_t ao = bb + kb * 1024 + lane * 32;
            v16bf a0 = *(const v16bf*)(smem + ao);            // h rt0
            v16bf a1 = *(const v16bf*)(smem + ao + 16384);    // h rt1
            long kk = c * KBC + kb;
            v16bf b = Bp[(nb + kk) * 32 + lane];
            acc0 = __builtin_amdgcn_wmma_f32_16x16x32_bf16(false, a0, false, b, (short)0, acc0, false, false);
            acc1 = __builtin_amdgcn_wmma_f32_16x16x32_bf16(false, a1, false, b, (short)0, acc1, false, false);
        }
        __syncthreads();
    }
    int col = c0 + (lane & 15);
    float bias = b_p[col];
#pragma unroll
    for (int i = 0; i < 8; ++i) {
        int mrow = i + ((lane >> 4) << 3);
        float v0 = acc0[i] + bias;
        float v1 = acc1[i] + bias;
        xhat[(long)mrow * V_ + col] = v0;
        xhat[(long)(16 + mrow) * V_ + col] = v1;
        if (outputs) {
            outputs[((long)mrow * S_ + t) * V_ + col] = v0;
            outputs[((long)(16 + mrow) * S_ + t) * V_ + col] = v1;
        }
    }
}

__global__ void copy_h(const float* __restrict__ src, float* __restrict__ dst) {
    int e = blockIdx.x * blockDim.x + threadIdx.x;
    dst[e] = src[e];
}

// ---------------------------------------------------------------------------
extern "C" void kernel_launch(void* const* d_in, const int* in_sizes, int n_in,
                              void* d_out, int out_size, void* d_ws, size_t ws_size,
                              hipStream_t stream) {
    const long long* x   = (const long long*)d_in[0];
    const float* h0      = (const float*)d_in[1];
    const float* W_ih    = (const float*)d_in[2];
    const float* W_hh    = (const float*)d_in[3];
    const float* b_ih    = (const float*)d_in[4];
    const float* b_hh    = (const float*)d_in[5];
    const float* W_p     = (const float*)d_in[6];
    const float* b_p     = (const float*)d_in[7];

    float* outputs = (float*)d_out;                       // (B,S,V)
    float* errors  = outputs + (long)B_ * S_ * V_;        // (B,S,V)
    float* h_final = errors  + (long)B_ * S_ * V_;        // (B,H)

    // workspace carve-up (~60 MB)
    char* ws = (char*)d_ws;
    uint16_t* Wih_p = (uint16_t*)ws;  ws += (long)3 * H_ * K_ * 2;
    uint16_t* Whh_p = (uint16_t*)ws;  ws += (long)3 * H_ * K_ * 2;
    uint16_t* Wp_p  = (uint16_t*)ws;  ws += (long)V_ * K_ * 2;
    float* h_f32[2];
    h_f32[0] = (float*)ws;            ws += (long)B_ * H_ * 4;
    h_f32[1] = (float*)ws;            ws += (long)B_ * H_ * 4;
    uint16_t* h_pk[2];
    h_pk[0] = (uint16_t*)ws;          ws += (long)B_ * H_ * 2;
    h_pk[1] = (uint16_t*)ws;          ws += (long)B_ * H_ * 2;
    uint16_t* e_pk = (uint16_t*)ws;   ws += (long)B_ * K_ * 2;
    float* xhat = (float*)ws;         ws += (long)B_ * V_ * 4;

    // 1) pack weights to bf16 fragment layout (stays hot in 192MB L2)
    {
        long n1 = (long)3 * H_ * K_;
        pack_w<<<(unsigned)((n1 + 255) / 256), 256, 0, stream>>>(W_ih, Wih_p, 3 * H_, K_);
        pack_w<<<(unsigned)((n1 + 255) / 256), 256, 0, stream>>>(W_hh, Whh_p, 3 * H_, K_);
        long n2 = (long)V_ * K_;
        pack_w<<<(unsigned)((n2 + 255) / 256), 256, 0, stream>>>(W_p, Wp_p, V_, K_);
    }
    // 2) initial state + x_hat0 = h0 @ W_p^T + b_p (not part of outputs)
    init_h<<<(B_ * H_ + 255) / 256, 256, 0, stream>>>(h0, h_f32[0], h_pk[0]);
    proj_step<<<V_ / NT, 128, 65536, stream>>>(h_pk[0], Wp_p, b_p, xhat, nullptr, 0);

    // 3) recurrence: kernel-launch boundaries are the inter-step barriers
    for (int t = 0; t < S_; ++t) {
        int p = t & 1;
        softmax_err<<<B_, 256, 0, stream>>>(xhat, x, t, errors, e_pk);
        gru_step<<<H_ / NT, 128, 131072, stream>>>(e_pk, h_pk[p], h_f32[p],
                                                   Wih_p, Whh_p, b_ih, b_hh,
                                                   h_f32[1 - p], h_pk[1 - p]);
        proj_step<<<V_ / NT, 128, 65536, stream>>>(h_pk[1 - p], Wp_p, b_p, xhat, outputs, t);
    }
    // last step (t=255, p=1) wrote h_f32[0]
    copy_h<<<(B_ * H_ + 255) / 256, 256, 0, stream>>>(h_f32[0], h_final);
}
